// SlidingWindowAttention_12764642804269
// MI455X (gfx1250) — matrix operand order repaired and checked by reading, not compiled, not run on hardware
//
#include <hip/hip_runtime.h>
#include <hip/hip_bf16.h>

// ---------------------------------------------------------------------------
// SlidingWindowAttention for MI455X (gfx1250, wave32, WMMA bf16 path)
// x[1,4096,2048] fp32 -> qkv proj -> per-head RMSNorm -> RoPE ->
// causal/window flash attention -> out proj -> fp32 [1,4096,2048]
// CDNA5 data movement: flash-attn K tile via Tensor Data Mover
// (tensor_load_to_lds + s_wait_tensorcnt, HW LDS padding), out-proj A tile
// via async global->LDS copy (global_load_async_to_lds_b128 + asynccnt).
// ---------------------------------------------------------------------------

typedef __bf16 bf16;
typedef bf16 v16bf  __attribute__((ext_vector_type(16)));
typedef bf16 bf16x8 __attribute__((ext_vector_type(8)));
typedef float v8f   __attribute__((ext_vector_type(8)));
typedef unsigned u32x4 __attribute__((ext_vector_type(4)));
typedef unsigned u32x8 __attribute__((ext_vector_type(8)));

constexpr int   N_TOK = 4096;
constexpr int   HDIM  = 2048;
constexpr int   NHEAD = 16;
constexpr int   HD    = 128;
constexpr int   WIN   = 4096;
constexpr float EPSV  = 1e-6f;
constexpr float NEGV  = -1e30f;
constexpr float INV_SQRT_HD = 0.08838834764831845f;   // 1/sqrt(128)
constexpr float LN_THETA_OVER_HALF = 0.14391156831212787f; // ln(10000)/64

#define DEVI __device__ __forceinline__

// Async 16-byte global->LDS copy (CDNA5, tracked by ASYNCcnt).
DEVI void async_copy_b128(const void* gsrc, void* lds_dst) {
  unsigned lds = (unsigned)(uintptr_t)lds_dst;
  unsigned long long ga = (unsigned long long)(uintptr_t)gsrc;
  asm volatile("global_load_async_to_lds_b128 %0, %1, off"
               :: "v"(lds), "v"(ga)
               : "memory");
}
DEVI void wait_asynccnt0() {
  asm volatile("s_wait_asynccnt 0" ::: "memory");
}

// Tensor Data Mover: DMA a 2D bf16 tile [rows x 128] from a [N_TOK x 128]
// head tensor into LDS, inserting 4 pad DWORDs after every 64 DWORDs so the
// LDS row stride is 136 bf16 (= our bank-conflict padding).  One instruction
// per wave; completion tracked by TENSORcnt.
DEVI void tdm_load_k_tile(const void* gsrc, void* lds_dst, int rows) {
  unsigned long long ga = (unsigned long long)(uintptr_t)gsrc;
  u32x4 g0;
  g0[0] = 1u;                                     // count=1, user descriptor
  g0[1] = (unsigned)(uintptr_t)lds_dst;           // lds_addr
  g0[2] = (unsigned)(ga & 0xFFFFFFFFu);           // global_addr[31:0]
  g0[3] = (unsigned)((ga >> 32) & 0x01FFFFFFu)    // global_addr[56:32]
          | 0x80000000u;                          // type=2 ("image")
  u32x8 g1;
  g1[0] = (1u << 16)     // data_size = 1 -> 2-byte elements
        | (1u << 20)     // pad_enable
        | (5u << 22)     // pad_interval: 64 DWORDs (one 128-elem bf16 row)
        | (3u << 25);    // pad_amount: 4 DWORDs (8 bf16) -> stride 136
  g1[1] = (unsigned)HD << 16;                     // tensor_dim0 = 128 (lo16)
  g1[2] = ((unsigned)N_TOK & 0xFFFFu) << 16;      // tensor_dim1 lo16
  g1[3] = ((unsigned)N_TOK >> 16)                 // tensor_dim1 hi16
        | ((unsigned)HD << 16);                   // tile_dim0 = 128
  g1[4] = (unsigned)rows;                         // tile_dim1 = rows, dim2=0
  g1[5] = (unsigned)HD;                           // tensor_dim0_stride lo32
  g1[6] = 0u;
  g1[7] = 0u;
  u32x4 z = {0u, 0u, 0u, 0u};
  asm volatile("tensor_load_to_lds %0, %1, %2, %3"
               :: "s"(g0), "s"(g1), "s"(z), "s"(z)
               : "memory");
}

// Build a v16bf from two contiguous 8-element (16-byte aligned) chunks.
DEVI v16bf make_frag(const bf16* p0, const bf16* p1) {
  bf16x8 lo = *(const bf16x8*)p0;
  bf16x8 hi = *(const bf16x8*)p1;
  v16bf r;
#pragma unroll
  for (int i = 0; i < 8; ++i) { r[i] = lo[i]; r[i + 8] = hi[i]; }
  return r;
}

// A-matrix fragment (16x32, row-major tile, stride in elements).
// ISA layout: lane<16 -> row=lane, K = {0..7, 16..23};
//             lane>=16 -> row=lane-16, K = {8..15, 24..31}.
DEVI v16bf load_a(const bf16* base, int stride, int lane) {
  int r  = lane & 15;
  int k0 = (lane < 16) ? 0 : 8;
  const bf16* p = base + r * stride + k0;
  return make_frag(p, p + 16);
}

// B-matrix fragment (32x16, tile stored K-contiguous per column: bt[col][k]).
// ISA layout: lane<16 -> col=lane, K = {0..7, 16..23};
//             lane>=16 -> col=lane-16, K = {8..15, 24..31}.
DEVI v16bf load_b(const bf16* base, int stride, int lane) {
  int c  = lane & 15;
  int k0 = (lane < 16) ? 0 : 8;
  const bf16* p = base + c * stride + k0;
  return make_frag(p, p + 16);
}

DEVI v8f wmma_bf16(v16bf a, v16bf b, v8f c) {
  return __builtin_amdgcn_wmma_f32_16x16x32_bf16(
      /*neg_a=*/false, a, /*neg_b=*/false, b,
      /*c_mod=*/(short)0, c, /*reuse_a=*/false, /*reuse_b=*/false);
}

// ---------------------------------------------------------------------------
// Kernel 1: QKV GEMM (x @ w_qkv) with fused RMSNorm + RoPE epilogue.
// ---------------------------------------------------------------------------
__global__ __launch_bounds__(256, 1)
void qkv_rmsnorm_rope_kernel(const float* __restrict__ x,
                             const float* __restrict__ w_qkv,
                             const float* __restrict__ qg,
                             const float* __restrict__ kg,
                             bf16* __restrict__ qbuf,
                             bf16* __restrict__ kbuf,
                             bf16* __restrict__ vbuf) {
  __shared__ bf16 As[64][40];     // 64x32 A tile (+8 pad -> 16B aligned rows)
  __shared__ bf16 Bt[128][40];    // 32x128 B tile, stored transposed [col][k]
  __shared__ float Cs[64][132];   // fp32 result staging for the epilogue

  const int cb   = blockIdx.x;          // 0..47  (type = cb/16, head = cb%16)
  const int rb   = blockIdx.y;          // 0..63
  const int tid  = threadIdx.x;
  const int lane = tid & 31;
  const int wv   = tid >> 5;
  const int wm   = wv & 3, wn = wv >> 2;

  v8f acc[4];
#pragma unroll
  for (int s = 0; s < 4; ++s)
#pragma unroll
    for (int e = 0; e < 8; ++e) acc[s][e] = 0.0f;

  for (int kt = 0; kt < HDIM / 32; ++kt) {
    // ---- stage A tile: 64x32 fp32 -> bf16 LDS ----
#pragma unroll
    for (int rep = 0; rep < 2; ++rep) {
      int idx = tid + rep * 256;            // 512 float4 slots
      int r = idx >> 3, c4 = idx & 7;
      const float4 f =
          *(const float4*)&x[(rb * 64 + r) * HDIM + kt * 32 + c4 * 4];
      bf16* d = &As[r][c4 * 4];
      d[0] = (bf16)f.x; d[1] = (bf16)f.y; d[2] = (bf16)f.z; d[3] = (bf16)f.w;
      if (kt + 1 < HDIM / 32)               // global_prefetch_b8 of next tile
        __builtin_prefetch(&x[(rb * 64 + r) * HDIM + (kt + 1) * 32 + c4 * 4], 0, 1);
    }
    // ---- stage B tile: 32x128 fp32 -> bf16 LDS, transposed [col][k] ----
#pragma unroll
    for (int rep = 0; rep < 4; ++rep) {
      int idx = tid + rep * 256;            // 1024 float4 slots
      int kr = idx >> 5, c4 = idx & 31;
      const float4 f = *(const float4*)
          &w_qkv[(kt * 32 + kr) * (3 * HDIM) + cb * 128 + c4 * 4];
      Bt[c4 * 4 + 0][kr] = (bf16)f.x;
      Bt[c4 * 4 + 1][kr] = (bf16)f.y;
      Bt[c4 * 4 + 2][kr] = (bf16)f.z;
      Bt[c4 * 4 + 3][kr] = (bf16)f.w;
    }
    __syncthreads();

    v16bf a = load_a(&As[16 * wm][0], 40, lane);
#pragma unroll
    for (int sub = 0; sub < 4; ++sub) {
      v16bf b = load_b(&Bt[64 * wn + 16 * sub][0], 40, lane);
      acc[sub] = wmma_bf16(a, b, acc[sub]);
    }
    __syncthreads();
  }

  // ---- spill accumulators to fp32 LDS tile (C layout: row=elem, col=lane) --
  const int hf = lane >> 4, ln = lane & 15;
#pragma unroll
  for (int sub = 0; sub < 4; ++sub)
#pragma unroll
    for (int e = 0; e < 8; ++e)
      Cs[16 * wm + 8 * hf + e][64 * wn + 16 * sub + ln] = acc[sub][e];
  __syncthreads();

  // ---- epilogue: 4 lanes per row, 32 columns each ----
  const int typ = cb >> 4, head = cb & 15;
  const int r = tid >> 2, c0 = (tid & 3) * 32;
  const int tok = rb * 64 + r;

  if (typ == 2) {                           // V: straight bf16 store
    bf16* dst = &vbuf[((size_t)head * N_TOK + tok) * HD];
#pragma unroll 8
    for (int c = c0; c < c0 + 32; ++c) dst[c] = (bf16)Cs[r][c];
  } else {                                  // Q/K: RMSNorm + RoPE
    float ssq = 0.0f;
#pragma unroll 8
    for (int c = c0; c < c0 + 32; ++c) { float v = Cs[r][c]; ssq += v * v; }
    ssq += __shfl_xor(ssq, 1, 32);
    ssq += __shfl_xor(ssq, 2, 32);
    const float rms = rsqrtf(ssq * (1.0f / HD) + EPSV);
    const float* g = (typ == 0) ? qg : kg;
    bf16* dst = ((typ == 0) ? qbuf : kbuf) + ((size_t)head * N_TOK + tok) * HD;
#pragma unroll 8
    for (int c = c0; c < c0 + 32; ++c) {
      float xn = Cs[r][c] * rms * g[c];
      int pc   = (c < 64) ? c + 64 : c - 64;
      float pn = Cs[r][pc] * rms * g[pc];
      float rh = (c < 64) ? -pn : pn;
      float ang = (float)tok * __expf((float)(c & 63) * -LN_THETA_OVER_HALF);
      float sn, cs; __sincosf(ang, &sn, &cs);
      dst[c] = (bf16)(xn * cs + rh * sn);
    }
  }
}

// ---------------------------------------------------------------------------
// Kernel 2: flash attention. grid = (N/128 query blocks, 16 heads),
// block = 256 threads = 8 waves; each wave owns 16 query rows for the whole
// kv loop. K tile is DMA'd by the Tensor Data Mover (wave 0 issues one
// descriptor; TDM applies the 136-element LDS row padding in hardware) while
// all waves stage the V^T tile through VGPRs, overlapping the DMA.
// ---------------------------------------------------------------------------
__global__ __launch_bounds__(256, 1)
void flash_attn_kernel(const bf16* __restrict__ qbuf,
                       const bf16* __restrict__ kbuf,
                       const bf16* __restrict__ vbuf,
                       bf16* __restrict__ obuf) {
  __shared__ bf16 Ks[64][136];    // K tile, row-major [kv][hd], TDM-padded
  __shared__ bf16 Vt[128][72];    // V tile transposed [hd][kv]
  __shared__ bf16 Ps[8][16][72];  // per-wave P staging (C-layout -> A-layout)

  const int qblk = blockIdx.x, head = blockIdx.y;
  const int q0   = qblk * 128;
  const int tid  = threadIdx.x, lane = tid & 31, wv = tid >> 5;
  const int hf   = lane >> 4, ln = lane & 15;

  // ---- load Q fragments (persist across the kv loop) ----
  v16bf qa[4];
  {
    const int row = q0 + 16 * wv + ln;
    const bf16* qp = qbuf + ((size_t)head * N_TOK + row) * HD;
#pragma unroll
    for (int ks = 0; ks < 4; ++ks) {
      const bf16* p = qp + 32 * ks + (hf ? 8 : 0);
      qa[ks] = make_frag(p, p + 16);
    }
  }

  v8f o[8];
  float m[8], l[8];
#pragma unroll
  for (int t = 0; t < 8; ++t)
#pragma unroll
    for (int e = 0; e < 8; ++e) o[t][e] = 0.0f;
#pragma unroll
  for (int e = 0; e < 8; ++e) { m[e] = NEGV; l[e] = 0.0f; }

  const int lo = max(0, q0 - WIN);
  for (int kv0 = lo & ~63; kv0 < q0 + 128; kv0 += 64) {
    __syncthreads();
    // ---- K tile: one TDM descriptor from wave 0 (DMA runs in background) --
    if (wv == 0)
      tdm_load_k_tile(&kbuf[((size_t)head * N_TOK + kv0) * HD], &Ks[0][0], 64);
    // ---- V^T tile: VGPR path (transpose), overlaps the TDM transfer ----
#pragma unroll
    for (int rep = 0; rep < 4; ++rep) {
      int idx = tid + rep * 256;           // 1024 chunks of 8 bf16
      int r = idx >> 4, c8 = idx & 15;
      const size_t gro = ((size_t)head * N_TOK + kv0 + r) * HD + c8 * 8;
      bf16x8 vvv = *(const bf16x8*)&vbuf[gro];
#pragma unroll
      for (int j = 0; j < 8; ++j) Vt[c8 * 8 + j][r] = vvv[j];
    }
    if (wv == 0) __builtin_amdgcn_s_wait_tensorcnt(0);
    __syncthreads();

    // ---- S = Q K^T / sqrt(HD), masked ----
    float pm[4][8];
    float mx[8];
#pragma unroll
    for (int e = 0; e < 8; ++e) mx[e] = NEGV;
#pragma unroll
    for (int sub = 0; sub < 4; ++sub) {
      v8f s;
#pragma unroll
      for (int e = 0; e < 8; ++e) s[e] = 0.0f;
#pragma unroll
      for (int ks = 0; ks < 4; ++ks) {
        v16bf b = load_b(&Ks[16 * sub][32 * ks], 136, lane);
        s = wmma_bf16(qa[ks], b, s);
      }
      const int j = kv0 + 16 * sub + ln;
#pragma unroll
      for (int e = 0; e < 8; ++e) {
        int qi = q0 + 16 * wv + 8 * hf + e;
        float sv = s[e] * INV_SQRT_HD;
        if (qi < j || qi - j > WIN) sv = NEGV;
        pm[sub][e] = sv;
        mx[e] = fmaxf(mx[e], sv);
      }
    }
    // row-max across the 16 lanes of each half-wave
#pragma unroll
    for (int e = 0; e < 8; ++e) {
      float v = mx[e];
      v = fmaxf(v, __shfl_xor(v, 1, 32));
      v = fmaxf(v, __shfl_xor(v, 2, 32));
      v = fmaxf(v, __shfl_xor(v, 4, 32));
      v = fmaxf(v, __shfl_xor(v, 8, 32));
      mx[e] = v;
    }
    // ---- online softmax update ----
    float alpha[8];
#pragma unroll
    for (int e = 0; e < 8; ++e) {
      float mn = fmaxf(m[e], mx[e]);
      alpha[e] = __expf(m[e] - mn);
      m[e] = mn;
      float sum = 0.0f;
#pragma unroll
      for (int sub = 0; sub < 4; ++sub) {
        // guard: fully-masked scores contribute exactly 0
        float p = (pm[sub][e] > 0.5f * NEGV) ? __expf(pm[sub][e] - mn) : 0.0f;
        pm[sub][e] = p;
        sum += p;
      }
      sum += __shfl_xor(sum, 1, 32);
      sum += __shfl_xor(sum, 2, 32);
      sum += __shfl_xor(sum, 4, 32);
      sum += __shfl_xor(sum, 8, 32);
      l[e] = l[e] * alpha[e] + sum;
    }
#pragma unroll
    for (int t = 0; t < 8; ++t)
#pragma unroll
      for (int e = 0; e < 8; ++e) o[t][e] *= alpha[e];

    // ---- transpose P through wave-private LDS (C-layout -> A-layout) ----
#pragma unroll
    for (int sub = 0; sub < 4; ++sub)
#pragma unroll
      for (int e = 0; e < 8; ++e)
        Ps[wv][8 * hf + e][16 * sub + ln] = (bf16)pm[sub][e];
    __builtin_amdgcn_wave_barrier();   // LDS ops from one wave are in-order

    // ---- O += P V ----
#pragma unroll
    for (int ks2 = 0; ks2 < 2; ++ks2) {
      v16bf a = load_a(&Ps[wv][0][32 * ks2], 72, lane);
#pragma unroll
      for (int t = 0; t < 8; ++t) {
        v16bf b = load_b(&Vt[16 * t][32 * ks2], 72, lane);
        o[t] = wmma_bf16(a, b, o[t]);
      }
    }
  }

  // ---- normalize and store: obuf[token][head*128 + hd], bf16 ----
#pragma unroll
  for (int e = 0; e < 8; ++e) l[e] = 1.0f / l[e];
  const int colb = head * HD;
#pragma unroll
  for (int t = 0; t < 8; ++t)
#pragma unroll
    for (int e = 0; e < 8; ++e) {
      int qi = q0 + 16 * wv + 8 * hf + e;
      obuf[(size_t)qi * HDIM + colb + 16 * t + ln] = (bf16)(o[t][e] * l[e]);
    }
}

// ---------------------------------------------------------------------------
// Kernel 3: out projection GEMM, attn[4096,2048](bf16) @ w_out[2048,2048](f32)
// -> d_out fp32. A tile is already bf16: staged via async global->LDS DMA.
// ---------------------------------------------------------------------------
__global__ __launch_bounds__(256, 1)
void out_proj_kernel(const bf16* __restrict__ ab,
                     const float* __restrict__ w_out,
                     float* __restrict__ out) {
  __shared__ bf16 As[64][40];
  __shared__ bf16 Bt[128][40];

  const int cb = blockIdx.x, rb = blockIdx.y;
  const int tid = threadIdx.x, lane = tid & 31, wv = tid >> 5;
  const int wm = wv & 3, wn = wv >> 2;

  v8f acc[4];
#pragma unroll
  for (int s = 0; s < 4; ++s)
#pragma unroll
    for (int e = 0; e < 8; ++e) acc[s][e] = 0.0f;

  for (int kt = 0; kt < HDIM / 32; ++kt) {
    {   // A: 64x32 bf16, 256 async 16B chunks (no conversion needed)
      int r = tid >> 2, c8 = tid & 3;
      const size_t gro = (size_t)(rb * 64 + r) * HDIM + kt * 32 + c8 * 8;
      async_copy_b128(&ab[gro], &As[r][c8 * 8]);
    }
    // B: fp32 -> bf16 conversion path
#pragma unroll
    for (int rep = 0; rep < 4; ++rep) {
      int idx = tid + rep * 256;
      int kr = idx >> 5, c4 = idx & 31;
      const float4 f =
          *(const float4*)&w_out[(kt * 32 + kr) * HDIM + cb * 128 + c4 * 4];
      Bt[c4 * 4 + 0][kr] = (bf16)f.x;
      Bt[c4 * 4 + 1][kr] = (bf16)f.y;
      Bt[c4 * 4 + 2][kr] = (bf16)f.z;
      Bt[c4 * 4 + 3][kr] = (bf16)f.w;
    }
    wait_asynccnt0();
    __syncthreads();

    v16bf a = load_a(&As[16 * wm][0], 40, lane);
#pragma unroll
    for (int sub = 0; sub < 4; ++sub) {
      v16bf b = load_b(&Bt[64 * wn + 16 * sub][0], 40, lane);
      acc[sub] = wmma_bf16(a, b, acc[sub]);
    }
    __syncthreads();
  }

  const int hf = lane >> 4, ln = lane & 15;
#pragma unroll
  for (int sub = 0; sub < 4; ++sub)
#pragma unroll
    for (int e = 0; e < 8; ++e) {
      int rg = rb * 64 + 16 * wm + 8 * hf + e;
      int cg = cb * 128 + 64 * wn + 16 * sub + ln;
      out[(size_t)rg * HDIM + cg] = acc[sub][e];
    }
}

// ---------------------------------------------------------------------------
extern "C" void kernel_launch(void* const* d_in, const int* in_sizes, int n_in,
                              void* d_out, int out_size, void* d_ws,
                              size_t ws_size, hipStream_t stream) {
  (void)in_sizes; (void)n_in; (void)out_size; (void)ws_size;
  const float* x     = (const float*)d_in[0];
  const float* w_qkv = (const float*)d_in[1];
  const float* w_out = (const float*)d_in[2];
  const float* qg    = (const float*)d_in[3];
  const float* kg    = (const float*)d_in[4];

  char* ws = (char*)d_ws;
  const size_t seg = (size_t)NHEAD * N_TOK * HD * sizeof(bf16);  // 16 MB
  bf16* qb = (bf16*)(ws + 0 * seg);
  bf16* kb = (bf16*)(ws + 1 * seg);
  bf16* vb = (bf16*)(ws + 2 * seg);
  bf16* ab = (bf16*)(ws + 3 * seg);

  qkv_rmsnorm_rope_kernel<<<dim3(3 * HDIM / 128, N_TOK / 64), 256, 0, stream>>>(
      x, w_qkv, qg, kg, qb, kb, vb);
  flash_attn_kernel<<<dim3(N_TOK / 128, NHEAD), 256, 0, stream>>>(
      qb, kb, vb, ab);
  out_proj_kernel<<<dim3(HDIM / 128, N_TOK / 64), 256, 0, stream>>>(
      ab, w_out, (float*)d_out);
}
